// BandSplit_42253888258230
// MI455X (gfx1250) — matrix operand order repaired and checked by reading, not compiled, not run on hardware
//
#include <hip/hip_runtime.h>
#include <hip/hip_bf16.h>

typedef __attribute__((ext_vector_type(16))) _Float16 v16h;
typedef __attribute__((ext_vector_type(8)))  _Float16 v8h;
typedef __attribute__((ext_vector_type(8)))  float    v8f;

#define BLOCK 256
#define N_BANDS 256
#define OUT_C 128
#define TOTAL_ROWS 1024   // B*T = 2*512
#define MTILE 128
#define F_BINS 1025
#define T_LEN 512
#define IN_C 4

__global__ void bs_zero_kernel(float* __restrict__ p, int n) {
    int i = blockIdx.x * blockDim.x + threadIdx.x;
    if (i < n) p[i] = 0.0f;
}

// Load a 16-element B-fragment column slice with affine stride (all loads
// unconditional & independent -> one wait, then convert+pack).
__device__ __forceinline__ v16h load_bfrag_affine(const float* __restrict__ base, int stride) {
    float w[16];
    #pragma unroll
    for (int j = 0; j < 16; j++) w[j] = base[(size_t)j * stride];
    v16h b;
    #pragma unroll
    for (int j = 0; j < 16; j++) b[j] = (_Float16)w[j];
    return b;
}

__device__ __forceinline__ v16h load_afrag_lds(const _Float16* __restrict__ rowp, int hl) {
    v8h a0 = *(const v8h*)(rowp + hl * 16);
    v8h a1 = *(const v8h*)(rowp + hl * 16 + 8);
    return __builtin_shufflevector(a0, a1, 0,1,2,3,4,5,6,7,8,9,10,11,12,13,14,15);
}

__global__ __launch_bounds__(BLOCK)
void bandsplit_kernel(const float* __restrict__ x,
                      const float* __restrict__ W1,
                      const float* __restrict__ b1,
                      const float* __restrict__ W2,
                      const float* __restrict__ b2,
                      const int*   __restrict__ nz_idx,
                      const float* __restrict__ nz_mel,
                      const float* __restrict__ maskp,
                      const float* __restrict__ ola,
                      float* __restrict__ out,
                      int Wb)
{
    const int K1  = 4 * Wb;                 // inner dim of GEMM1 / outer of GEMM2
    const int Kp1 = (K1 + 31) & ~31;        // padded to WMMA K chunks
    const int SA  = Kp1 + 8;                // gather-tile row stride (halves)
    const int SH  = 136;                    // h-tile row stride (halves), 128+8

    const int band    = blockIdx.x;
    const int rowBase = blockIdx.y * MTILE;

    extern __shared__ char smem_raw[];
    _Float16* sMat = (_Float16*)smem_raw;   // union: gather tile / h tile
    const int matHalves = MTILE * (SA > SH ? SA : SH);
    int*   sIdx = (int*)(smem_raw + (((size_t)matHalves * 2 + 15) & ~(size_t)15));
    float* sMel = (float*)(sIdx + Wb);

    const int tid  = threadIdx.x;
    const int lane = tid & 31;
    const int wave = tid >> 5;
    const int col  = lane & 15;             // N / M index within fragment
    const int hl   = lane >> 4;             // half-wave select

    // ---- stage band metadata ----
    for (int w = tid; w < Wb; w += BLOCK) {
        sIdx[w] = nz_idx[band * Wb + w];
        sMel[w] = nz_mel[band * Wb + w];
    }
    __syncthreads();

    // ---- gather g tile: g[r][w*4+c] = x[b][c][t][idx[w]] * mel[w]  (f16) ----
    // Unconditional loads (clamped band-window index), select-to-zero in pad.
    for (int r = wave; r < MTILE; r += 8) {
        const int R  = rowBase + r;
        const int bb = R >> 9;              // / T_LEN
        const int tt = R & 511;
        const float* xrow = x + ((size_t)bb * IN_C * T_LEN + tt) * F_BINS;
        _Float16* rowp = sMat + r * SA;
        for (int i = lane; i < Kp1; i += 32) {
            const int wi = i >> 2, c = i & 3;
            const int wc = wi < Wb - 1 ? wi : Wb - 1;
            const int f  = sIdx[wc];
            const float melw = sMel[wc];
            const float xv = xrow[(size_t)c * T_LEN * F_BINS + f];
            const float v  = (i < K1) ? xv * melw : 0.0f;
            rowp[i] = (_Float16)v;
        }
    }
    __syncthreads();

    // ---- GEMM1: (128 x K1) @ (K1 x 128); wave owns N-tile n0, all 8 M-tiles ----
    v8f acc[8];
    #pragma unroll
    for (int mt = 0; mt < 8; mt++) acc[mt] = (v8f){0.f,0.f,0.f,0.f,0.f,0.f,0.f,0.f};

    const int n0 = wave * 16;
    const float* W1k = W1 + (size_t)band * K1 * OUT_C;
    const int Kfull = K1 & ~31;             // complete 32-chunks (no bounds checks)

    int ks = 0;
    for (; ks < Kfull; ks += 32) {
        v16h bfrag = load_bfrag_affine(W1k + (size_t)(ks + hl * 16) * OUT_C + n0 + col, OUT_C);
        #pragma unroll
        for (int mt = 0; mt < 8; mt++) {
            v16h afrag = load_afrag_lds(sMat + (mt * 16 + col) * SA + ks, hl);
            acc[mt] = __builtin_amdgcn_wmma_f32_16x16x32_f16(
                false, afrag, false, bfrag, (short)0, acc[mt], false, false);
        }
    }
    if (ks < Kp1) {
        // Tail chunk: clamp K index (A is exactly zero in the LDS pad, so the
        // clamped finite weights contribute 0).
        float w[16];
        #pragma unroll
        for (int j = 0; j < 16; j++) {
            int kidx = ks + hl * 16 + j;
            kidx = kidx < K1 - 1 ? kidx : K1 - 1;
            w[j] = W1k[(size_t)kidx * OUT_C + n0 + col];
        }
        v16h bfrag;
        #pragma unroll
        for (int j = 0; j < 16; j++) bfrag[j] = (_Float16)w[j];
        #pragma unroll
        for (int mt = 0; mt < 8; mt++) {
            v16h afrag = load_afrag_lds(sMat + (mt * 16 + col) * SA + ks, hl);
            acc[mt] = __builtin_amdgcn_wmma_f32_16x16x32_f16(
                false, afrag, false, bfrag, (short)0, acc[mt], false, false);
        }
    }
    __syncthreads();   // all waves done reading gather tile

    // ---- h = acc + b1 -> LDS (f16), stride SH ----
    const float b1o = b1[band * OUT_C + n0 + col];
    #pragma unroll
    for (int mt = 0; mt < 8; mt++) {
        #pragma unroll
        for (int v = 0; v < 8; v++) {
            const int m = mt * 16 + v + hl * 8;
            sMat[m * SH + n0 + col] = (_Float16)(acc[mt][v] + b1o);
        }
    }
    __syncthreads();

    // ---- GEMM2: (128 x 128) @ (128 x K1); waves stripe output tile columns ----
    const int NT2 = (K1 + 15) >> 4;
    const float* W2k = W2 + (size_t)band * OUT_C * K1;
    for (int nt = wave; nt < NT2; nt += 8) {
        const int o  = nt * 16 + col;
        const bool ov = (o < K1);
        // Clamped column base: out-of-range columns only pollute D elements in
        // lanes that skip the epilogue write, so no select is needed.
        const float* W2col = W2k + (ov ? o : K1 - 1);

        v8f acc2[8];
        #pragma unroll
        for (int mt = 0; mt < 8; mt++) acc2[mt] = (v8f){0.f,0.f,0.f,0.f,0.f,0.f,0.f,0.f};

        #pragma unroll
        for (int ks2 = 0; ks2 < 128; ks2 += 32) {
            v16h bfrag = load_bfrag_affine(W2col + (size_t)(ks2 + hl * 16) * K1, K1);
            #pragma unroll
            for (int mt = 0; mt < 8; mt++) {
                v16h afrag = load_afrag_lds(sMat + (mt * 16 + col) * SH + ks2, hl);
                acc2[mt] = __builtin_amdgcn_wmma_f32_16x16x32_f16(
                    false, afrag, false, bfrag, (short)0, acc2[mt], false, false);
            }
        }

        // ---- epilogue: (y + b2) * mask / ola[idx] scatter-add ----
        if (ov) {
            const int w = o >> 2, c = o & 3;
            const float mw = maskp[band * Wb + w];
            if (mw != 0.0f) {
                const int f = sIdx[w];
                const float s   = mw / ola[f];
                const float b2o = b2[(size_t)band * K1 + o];
                #pragma unroll
                for (int mt = 0; mt < 8; mt++) {
                    #pragma unroll
                    for (int v = 0; v < 8; v++) {
                        const int R  = rowBase + mt * 16 + v + hl * 8;
                        const int bi = R >> 9;
                        const int ti = R & 511;
                        const float val = (acc2[mt][v] + b2o) * s;
                        unsafeAtomicAdd(out + (((size_t)(bi * IN_C + c) * T_LEN + ti) * F_BINS + f), val);
                    }
                }
            }
        }
    }
}

extern "C" void kernel_launch(void* const* d_in, const int* in_sizes, int n_in,
                              void* d_out, int out_size, void* d_ws, size_t ws_size,
                              hipStream_t stream) {
    (void)n_in; (void)d_ws; (void)ws_size;
    const float* x    = (const float*)d_in[0];
    const float* W1   = (const float*)d_in[1];
    const float* b1   = (const float*)d_in[2];
    const float* W2   = (const float*)d_in[3];
    const float* b2   = (const float*)d_in[4];
    const int*   idx  = (const int*)  d_in[5];
    const float* mel  = (const float*)d_in[6];
    const float* mask = (const float*)d_in[7];
    const float* ola  = (const float*)d_in[8];
    float* out = (float*)d_out;

    const int Wb  = in_sizes[5] / N_BANDS;
    const int K1  = 4 * Wb;
    const int Kp1 = (K1 + 31) & ~31;
    const int SA  = Kp1 + 8;
    const int SH  = 136;
    const int matHalves = MTILE * (SA > SH ? SA : SH);
    size_t smem = (((size_t)matHalves * 2 + 15) & ~(size_t)15)
                + (size_t)Wb * 4   /* sIdx */
                + (size_t)Wb * 4   /* sMel */
                + 64;

    bs_zero_kernel<<<(out_size + BLOCK - 1) / BLOCK, BLOCK, 0, stream>>>(out, out_size);

    dim3 grid(N_BANDS, TOTAL_ROWS / MTILE);   // 256 bands x 8 row tiles
    bandsplit_kernel<<<grid, BLOCK, smem, stream>>>(
        x, W1, b1, W2, b2, idx, mel, mask, ola, out, Wb);
}
